// SuperPointInterestPointDecoder_5823975653807
// MI455X (gfx1250) — matrix operand order repaired and checked by reading, not compiled, not run on hardware
//
#include <hip/hip_runtime.h>

typedef __attribute__((ext_vector_type(16))) _Float16 v16h;
typedef __attribute__((ext_vector_type(8)))  _Float16 v8h;   // 16B, align 16
typedef __attribute__((ext_vector_type(8)))  float    v8f;
typedef __attribute__((ext_vector_type(4)))  unsigned int u32x4;
typedef __attribute__((ext_vector_type(8)))  int      i32x8;
typedef __attribute__((ext_vector_type(4)))  int      i32x4;

#if defined(__has_builtin)
#if __has_builtin(__builtin_amdgcn_tensor_load_to_lds) && \
    __has_builtin(__builtin_amdgcn_s_wait_tensorcnt)
#define HAVE_TDM 1
#endif
#endif
#ifndef HAVE_TDM
#define HAVE_TDM 0
#endif

union Frag  { v16h v; v8h h[2]; };
union Pack8 { v8h v; unsigned short u[8]; };

__device__ __forceinline__ unsigned short f2h(float f) {
    _Float16 h = (_Float16)f;
    return __builtin_bit_cast(unsigned short, h);
}

#define NPIX (4096u * 4096u)
#define CAND_CAP 8192

// ---------------------------------------------------------------------------
// Weight repack: f32 -> f16, GEMM-ready, bank-swizzled layouts.
//  w1p: per K-step (36 steps of 32) transposed tiles: [ks][n=256][kl=32],
//       8-elem block index XORed with (n>>2)&3.   k = (ky*3+kx)*128 + c
//  w2p: [n=80][k=256], block index XORed with n&15; n>=65 zero-padded.
// ---------------------------------------------------------------------------
__global__ __launch_bounds__(256) void repack_weights(
    const float* __restrict__ w1, const float* __restrict__ w2,
    unsigned short* __restrict__ w1p, unsigned short* __restrict__ w2p)
{
    int i = blockIdx.x * 256 + threadIdx.x;
    if (i < 1152 * 256) {
        int k = i / 256, n = i % 256;
        int c = k & 127, p = k >> 7;            // p = ky*3+kx
        int ks = k >> 5, kl = k & 31;
        int dst = ks * 8192 + n * 32 + (((kl >> 3) ^ ((n >> 2) & 3)) << 3) + (kl & 7);
        w1p[dst] = f2h(w1[(size_t)n * 1152 + c * 9 + p]);
    } else {
        int j = i - 1152 * 256;
        if (j < 80 * 256) {
            int n = j / 256, k = j % 256;
            int dst = n * 256 + (((k >> 3) ^ (n & 15)) << 3) + (k & 7);
            w2p[dst] = f2h(n < 65 ? w2[(size_t)n * 256 + k] : 0.0f);
        }
    }
}

// ---------------------------------------------------------------------------
// Fused head: conv3x3(128->256)+ReLU -> conv1x1(256->65) -> softmax(65)
//             -> depth-to-space 8x into S[4096][4096].
// Grid: (8 x-tiles of 64 px, 512 rows). Block: 256 threads = 8 wave32.
// GEMM1: M=64,N=256,K=1152 (f16 WMMA, f32 acc), double-buffered K-steps.
// GEMM2: M=64,N=80(65 valid),K=256 from LDS; W2 tile arrives via TDM DMA
//        overlapped with all of GEMM1.
// ---------------------------------------------------------------------------
__global__ __launch_bounds__(256) void head_fused(
    const float* __restrict__ x, const float* __restrict__ b1,
    const float* __restrict__ b2, const unsigned short* __restrict__ w1p,
    const unsigned short* __restrict__ w2p, float* __restrict__ S)
{
    __shared__ __align__(32) unsigned char smem[114688];
    unsigned short* Ab  = (unsigned short*)smem;            // 2 x [m=64][k=32]
    unsigned short* Bb  = (unsigned short*)(smem + 8192);   // 2 x [n=256][k=32]
    float*          Sm  = (float*)smem;                     // 64x80, reuses A/B
    unsigned short* Mid = (unsigned short*)(smem + 40960);  // [m=64][k=256]
    unsigned short* W2T = (unsigned short*)(smem + 73728);  // [n=80][k=256]

    const int tid = threadIdx.x;
    const int x0  = blockIdx.x * 64;
    const int y   = blockIdx.y;

    const int w    = tid >> 5;       // wave32
    const int l    = tid & 31;
    const int lm   = l & 15;
    const int hf   = l >> 4;         // 0/1: K-quadrant half of the lane
    const int mt   = w >> 1;         // M tile 0..3
    const int nh   = w & 1;          // N half 0..1
    const int sA   = (lm >> 2) & 3;  // A swizzle for this lane's row

#if HAVE_TDM
    // Tensor Data Mover: DMA the 40KB W2 tile (linear, swizzle pre-baked in
    // w2p) into LDS while GEMM1 runs. 1-D tile: 5120 x 8-byte elements.
    if (w == 0) {
        unsigned long long ga = (unsigned long long)w2p;
        unsigned ldsOff = (unsigned)(unsigned long long)(void*)W2T;
        u32x4 g0;
        g0[0] = 1u;                                         // count=1, user D#
        g0[1] = ldsOff;                                     // lds_addr (bytes)
        g0[2] = (unsigned)ga;                               // global_addr[31:0]
        g0[3] = (unsigned)((ga >> 32) & 0x1FFFFFFu)         // global_addr[56:32]
              | (2u << 30);                                 // type=2 ("image")
        i32x8 g1;
        g1[0] = 0x00030000;            // wg_mask=0, data_size=3 (8B), no pad/iter
        g1[1] = (int)(5120u << 16);    // tensor_dim0[15:0] in bits [63:48]
        g1[2] = (int)(1u << 16);       // tensor_dim0 hi=0; tensor_dim1=1
        g1[3] = (int)(5120u << 16);    // tile_dim0 in bits [127:112]
        g1[4] = 0;                     // tile_dim1=0 (1-D), tile_dim2=0
        g1[5] = 5120;                  // tensor_dim0_stride[31:0]
        g1[6] = 0;
        g1[7] = 0;
        i32x4 z4 = {0, 0, 0, 0};
        i32x8 z8 = {0, 0, 0, 0, 0, 0, 0, 0};
        __builtin_amdgcn_tensor_load_to_lds(g0, g1, z4, z4, z8, 0);
    }
#else
    { // fallback: vectorized copy (swizzle pre-baked): straight b128 copy
        const uint4* src = (const uint4*)w2p;
        uint4* dst = (uint4*)W2T;
        for (int i = tid; i < 2560; i += 256) dst[i] = src[i];
    }
#endif

    // im2col loader mapping: lane-contiguous in m for coalesced x reads
    const int mL   = tid & 63;
    const int kg   = tid >> 6;                       // 8-k block 0..3
    const int aStB = ((kg ^ ((mL >> 2) & 3)) << 3);  // swizzled store block

    v8f zero8 = {};
    v8f acc[8];
#pragma unroll
    for (int i = 0; i < 8; ++i) acc[i] = zero8;

    float areg[8];
    uint4 breg[4];

    auto issueLoads = [&](int ks) {
#pragma unroll
        for (int kk = 0; kk < 8; ++kk) {
            int k  = ks * 32 + kg * 8 + kk;
            int c  = k & 127;
            int p  = k >> 7;
            int ky = p / 3;
            int kx = p - ky * 3;
            int yy = y + ky - 1;
            int xx = x0 + mL + kx - 1;
            areg[kk] = ((unsigned)yy < 512u && (unsigned)xx < 512u)
                     ? x[(size_t)c * 262144 + (size_t)yy * 512 + xx] : 0.0f;
        }
        const uint4* src = (const uint4*)(w1p + (size_t)ks * 8192);
#pragma unroll
        for (int i = 0; i < 4; ++i) breg[i] = src[tid + i * 256];
        if (ks + 1 < 36) __builtin_prefetch(w1p + (size_t)(ks + 1) * 8192, 0, 3);
    };

    auto storeTiles = [&](int buf) {
        Pack8 pk;
#pragma unroll
        for (int kk = 0; kk < 8; ++kk) pk.u[kk] = f2h(areg[kk]);
        *(v8h*)(Ab + buf * 2048 + mL * 32 + aStB) = pk.v;
        uint4* dst = (uint4*)(Bb + buf * 8192);
#pragma unroll
        for (int i = 0; i < 4; ++i) dst[tid + i * 256] = breg[i];
    };

    issueLoads(0);
    storeTiles(0);
    __syncthreads();

    for (int ks = 0; ks < 36; ++ks) {
        if (ks + 1 < 36) issueLoads(ks + 1);

        const unsigned short* A = Ab + (ks & 1) * 2048;
        const unsigned short* B = Bb + (ks & 1) * 8192;
        Frag a;
        {   // A frag: blocks hf and hf|2, swizzled by row
            const unsigned short* ap = A + (mt * 16 + lm) * 32;
            a.h[0] = *(const v8h*)(ap + ((hf       ^ sA) << 3));
            a.h[1] = *(const v8h*)(ap + (((hf | 2) ^ sA) << 3));
        }
#pragma unroll
        for (int nt = 0; nt < 8; ++nt) {
            int n = (nh * 8 + nt) * 16 + lm;
            const unsigned short* bp = B + n * 32;
            int sB = (n >> 2) & 3;
            int b0 = hf;                       // blocks hf, hf|2
            Frag b;
            b.h[0] = *(const v8h*)(bp + ((b0       ^ sB) << 3));
            b.h[1] = *(const v8h*)(bp + (((b0 | 2) ^ sB) << 3));
            acc[nt] = __builtin_amdgcn_wmma_f32_16x16x32_f16(
                false, a.v, false, b.v, (short)0, acc[nt], false, false);
        }

        if (ks + 1 < 36) storeTiles((ks + 1) & 1);
        __syncthreads();
    }

    // bias + ReLU -> Mid (f16, swizzled by (m&15))
#pragma unroll
    for (int nt = 0; nt < 8; ++nt) {
        int nMid = (nh * 8 + nt) * 16 + lm;
        int nb = nMid >> 3, nl = nMid & 7;
        float bias = b1[nMid];
#pragma unroll
        for (int r = 0; r < 8; ++r) {
            int mr  = hf * 8 + r;              // m & 15
            int m   = mt * 16 + mr;
            float v = acc[nt][r] + bias;
            Mid[m * 256 + ((nb ^ mr) << 3) + nl] = f2h(v > 0.0f ? v : 0.0f);
        }
    }
#if HAVE_TDM
    __builtin_amdgcn_s_wait_tensorcnt(0);   // wave 0 drains the W2 DMA
#endif
    __syncthreads();

    // GEMM2: 20 tiles (4 M x 5 N) round-robin over 8 waves
    for (int t = w; t < 20; t += 8) {
        int m2 = t & 3;
        int n2 = t >> 2;
        const unsigned short* ap = Mid + (m2 * 16 + lm) * 256;
        const unsigned short* bp = W2T + (n2 * 16 + lm) * 256;
        v8f acc2 = zero8;
#pragma unroll
        for (int ks = 0; ks < 8; ++ks) {
            Frag a, b;
            int a0 = (ks * 4 + hf)            ^ lm;   // Mid swizzle uses m&15==lm
            int a1 = (ks * 4 + hf + 2)        ^ lm;
            a.h[0] = *(const v8h*)(ap + (a0 << 3));
            a.h[1] = *(const v8h*)(ap + (a1 << 3));
            int c0 = (ks * 4 + hf)            ^ lm;   // W2T swizzle uses n&15==lm
            int c1 = (ks * 4 + hf + 2)        ^ lm;
            b.h[0] = *(const v8h*)(bp + (c0 << 3));
            b.h[1] = *(const v8h*)(bp + (c1 << 3));
            acc2 = __builtin_amdgcn_wmma_f32_16x16x32_f16(
                false, a.v, false, b.v, (short)0, acc2, false, false);
        }
        int n = n2 * 16 + lm;
        float bias = (n < 65) ? b2[n] : 0.0f;
#pragma unroll
        for (int r = 0; r < 8; ++r) {
            int m = m2 * 16 + hf * 8 + r;
            Sm[m * 80 + n] = acc2[r] + bias;
        }
    }
    __syncthreads();

    // softmax over 65 logits, drop dustbin, depth-to-space 8x8 scatter
    if (tid < 64) {
        const float* L = Sm + tid * 80;
        float mx = L[0];
        for (int c = 1; c < 65; ++c) mx = fmaxf(mx, L[c]);
        float s = 0.0f;
        for (int c = 0; c < 65; ++c) s += __expf(L[c] - mx);
        float inv = 1.0f / s;
        int xg = x0 + tid;
        for (int c = 0; c < 64; ++c) {
            float e = __expf(L[c] - mx) * inv;
            S[(size_t)(y * 8 + (c >> 3)) * 4096 + (size_t)xg * 8 + (c & 7)] = e;
        }
    }
}

// ---------------------------------------------------------------------------
// Separable 9-wide max pools (SAME with -inf pad == clamped window)
// ---------------------------------------------------------------------------
__global__ __launch_bounds__(256) void rowmax9(const float* __restrict__ in,
                                               float* __restrict__ out)
{
    unsigned i = blockIdx.x * 256u + threadIdx.x;
    int y = i >> 12, x = i & 4095;
    const float* row = in + ((size_t)y << 12);
    int lo = x - 4 < 0 ? 0 : x - 4;
    int hi = x + 4 > 4095 ? 4095 : x + 4;
    float m = row[lo];
    for (int xx = lo + 1; xx <= hi; ++xx) m = fmaxf(m, row[xx]);
    out[i] = m;
}

__global__ __launch_bounds__(256) void colmax_mask0(const float* __restrict__ S,
                                                    const float* __restrict__ R,
                                                    float* __restrict__ M)
{
    unsigned i = blockIdx.x * 256u + threadIdx.x;
    int y = i >> 12, x = i & 4095;
    int lo = y - 4 < 0 ? 0 : y - 4;
    int hi = y + 4 > 4095 ? 4095 : y + 4;
    float m = R[((size_t)lo << 12) + x];
    for (int yy = lo + 1; yy <= hi; ++yy) m = fmaxf(m, R[((size_t)yy << 12) + x]);
    M[i] = (S[i] == m) ? 1.0f : 0.0f;
}

// pooled(mask) > 0 -> suppressed; encode supp as -1 (scores are >= 0)
__global__ __launch_bounds__(256) void colmax_supp(const float* __restrict__ R,
                                                   const float* __restrict__ S,
                                                   float* __restrict__ SS)
{
    unsigned i = blockIdx.x * 256u + threadIdx.x;
    int y = i >> 12, x = i & 4095;
    int lo = y - 4 < 0 ? 0 : y - 4;
    int hi = y + 4 > 4095 ? 4095 : y + 4;
    float m = R[((size_t)lo << 12) + x];
    for (int yy = lo + 1; yy <= hi; ++yy) m = fmaxf(m, R[((size_t)yy << 12) + x]);
    SS[i] = (m > 0.0f) ? -1.0f : S[i];
}

// mask |= (supp_scores == pool(supp_scores)) & ~supp
__global__ __launch_bounds__(256) void colmax_update(const float* __restrict__ R,
                                                     const float* __restrict__ SS,
                                                     float* __restrict__ M)
{
    unsigned i = blockIdx.x * 256u + threadIdx.x;
    int y = i >> 12, x = i & 4095;
    int lo = y - 4 < 0 ? 0 : y - 4;
    int hi = y + 4 > 4095 ? 4095 : y + 4;
    float m = R[((size_t)lo << 12) + x];
    for (int yy = lo + 1; yy <= hi; ++yy) m = fmaxf(m, R[((size_t)yy << 12) + x]);
    float ss = SS[i];
    if (ss >= 0.0f && ss == m) M[i] = 1.0f;
}

// ---------------------------------------------------------------------------
// Top-k via 16-bit-key histogram select + bitonic sort of survivors
// ---------------------------------------------------------------------------
__device__ __forceinline__ bool valid_score(const float* S, const float* M,
                                            unsigned i, float* sf, unsigned* key)
{
    int y = i >> 12, x = i & 4095;
    float v = (M[i] > 0.0f) ? S[i] : 0.0f;
    *sf = v;
    bool ok = (v > 0.005f) && x >= 4 && x < 4092 && y >= 4 && y < 4092;
    *key = __float_as_uint(v) >> 15;   // monotonic for positive floats, < 65536
    return ok;
}

__global__ __launch_bounds__(256) void zero_small(unsigned* __restrict__ hist,
                                                  unsigned* __restrict__ counters)
{
    int i = blockIdx.x * 256 + threadIdx.x;
    if (i < 65536) hist[i] = 0u;
    else if (i < 65536 + 64) counters[i - 65536] = 0u;
}

__global__ __launch_bounds__(256) void hist_build(const float* __restrict__ S,
                                                  const float* __restrict__ M,
                                                  unsigned* __restrict__ hist)
{
    unsigned i = blockIdx.x * 256u + threadIdx.x;
    float sf; unsigned key;
    if (valid_score(S, M, i, &sf, &key)) atomicAdd(&hist[key], 1u);
}

__global__ void select_cutoff(const unsigned* __restrict__ hist,
                              unsigned* __restrict__ counters)
{
    if (threadIdx.x == 0 && blockIdx.x == 0) {
        unsigned long long c = 0;
        int T = 0;
        for (int k = 65535; k >= 0; --k) {
            c += hist[k];
            if (c >= 1024ull) { T = k; break; }
        }
        while (c > (unsigned long long)CAND_CAP && T < 65536) {
            c -= hist[T];
            ++T;
        }
        counters[1] = (unsigned)T;
    }
}

__global__ __launch_bounds__(256) void compact_cands(const float* __restrict__ S,
                                                     const float* __restrict__ M,
                                                     unsigned* __restrict__ counters,
                                                     float* __restrict__ candS,
                                                     int* __restrict__ candI)
{
    unsigned i = blockIdx.x * 256u + threadIdx.x;
    unsigned T = counters[1];
    float sf; unsigned key;
    if (valid_score(S, M, i, &sf, &key) && key >= T) {
        unsigned pos = atomicAdd(&counters[0], 1u);
        if (pos < CAND_CAP) { candS[pos] = sf; candI[pos] = (int)i; }
    }
}

__global__ __launch_bounds__(1024) void sort_emit(const float* __restrict__ candS,
                                                  const int* __restrict__ candI,
                                                  const unsigned* __restrict__ counters,
                                                  float* __restrict__ out)
{
    __shared__ float s[CAND_CAP];
    __shared__ int   id[CAND_CAP];
    unsigned n = counters[0];
    if (n > CAND_CAP) n = CAND_CAP;
    for (int i = threadIdx.x; i < CAND_CAP; i += 1024) {
        if ((unsigned)i < n) { s[i] = candS[i]; id[i] = candI[i]; }
        else                 { s[i] = -3.0e38f; id[i] = 0x7FFFFFFF; }
    }
    __syncthreads();
    // bitonic sort: score desc, index asc (deterministic despite atomic order)
    for (int k = 2; k <= CAND_CAP; k <<= 1) {
        for (int j = k >> 1; j > 0; j >>= 1) {
            for (int i = threadIdx.x; i < CAND_CAP; i += 1024) {
                int p = i ^ j;
                if (p > i) {
                    float si = s[i], sp = s[p];
                    int   ii = id[i], ip = id[p];
                    bool iFirst = (si > sp) || (si == sp && ii < ip);
                    bool desc = ((i & k) == 0);
                    if (desc ? !iFirst : iFirst) {
                        s[i] = sp; s[p] = si; id[i] = ip; id[p] = ii;
                    }
                }
            }
            __syncthreads();
        }
    }
    {
        int jj = threadIdx.x;
        float sc, kxf, kyf;
        if ((unsigned)jj < n) {
            sc = s[jj];
            int idx = id[jj];
            kxf = (float)(idx & 4095);
            kyf = (float)(idx >> 12);
        } else {
            // mimic top_k over -1-filled flat array: earliest flat indices win
            int idx = jj - (int)n;
            sc  = -1.0f;
            kxf = (float)(idx & 4095);
            kyf = (float)(idx >> 12);
        }
        out[jj * 2 + 0] = kxf;
        out[jj * 2 + 1] = kyf;
        out[2048 + jj]  = sc;
    }
}

// ---------------------------------------------------------------------------
extern "C" void kernel_launch(void* const* d_in, const int* in_sizes, int n_in,
                              void* d_out, int out_size, void* d_ws, size_t ws_size,
                              hipStream_t stream)
{
    (void)in_sizes; (void)n_in; (void)out_size; (void)ws_size;
    const float* x  = (const float*)d_in[0];   // [1,128,512,512]
    const float* w1 = (const float*)d_in[1];   // [256,128,3,3]
    const float* b1 = (const float*)d_in[2];   // [256]
    const float* w2 = (const float*)d_in[3];   // [65,256,1,1]
    const float* b2 = (const float*)d_in[4];   // [65]
    float* out = (float*)d_out;                // 1024*2 keypoints + 1024 scores

    char* ws = (char*)d_ws;
    float* S  = (float*)ws;                    // score map 4096^2
    float* R  = S + NPIX;                      // separable pool temp
    float* M  = R + NPIX;                      // NMS mask (0/1 float)
    float* SS = M + NPIX;                      // suppressed scores (-1 = supp)
    unsigned short* w1p = (unsigned short*)(SS + NPIX);      // 36x256x32 f16
    unsigned short* w2p = w1p + 1152 * 256;                  // 80x256 f16
    unsigned* hist     = (unsigned*)(w2p + 80 * 256);        // 65536 bins
    unsigned* counters = hist + 65536;                       // [0]=cnt [1]=cutoff
    float* candS = (float*)(counters + 64);
    int*   candI = (int*)(candS + CAND_CAP);

    repack_weights<<<1233, 256, 0, stream>>>(w1, w2, w1p, w2p);
    head_fused<<<dim3(8, 512), 256, 0, stream>>>(x, b1, b2, w1p, w2p, S);

    rowmax9<<<65536, 256, 0, stream>>>(S, R);
    colmax_mask0<<<65536, 256, 0, stream>>>(S, R, M);
    for (int it = 0; it < 2; ++it) {
        rowmax9<<<65536, 256, 0, stream>>>(M, R);
        colmax_supp<<<65536, 256, 0, stream>>>(R, S, SS);
        rowmax9<<<65536, 256, 0, stream>>>(SS, R);
        colmax_update<<<65536, 256, 0, stream>>>(R, SS, M);
    }

    zero_small<<<257, 256, 0, stream>>>(hist, counters);
    hist_build<<<65536, 256, 0, stream>>>(S, M, hist);
    select_cutoff<<<1, 32, 0, stream>>>(hist, counters);
    compact_cands<<<65536, 256, 0, stream>>>(S, M, counters, candS, candI);
    sort_emit<<<1, 1024, 0, stream>>>(candS, candI, counters, out);
}
